// GCNNClassifier_343597384055
// MI455X (gfx1250) — matrix operand order repaired and checked by reading, not compiled, not run on hardware
//
#include <hip/hip_runtime.h>
#include <math.h>

typedef __attribute__((ext_vector_type(16))) _Float16 v16h;
typedef __attribute__((ext_vector_type(8)))  float    v8f;

// ---------------------------------------------------------------------------
// Pack W[K,M] (f32, row-major) into WMMA B-fragment layout (f16):
//   Wp[kstep][colTile][lane][e]  with  k = kstep*32 + (lane>=16?16:0) + e,
//                                      n = colTile*16 + (lane&15)
// Rows K..Kpad-1 are zero-filled so the GEMM needs no k-guards.
// ---------------------------------------------------------------------------
__global__ void k_pack_w(const float* __restrict__ W, _Float16* __restrict__ Wp,
                         int K, int Kpad, int M) {
  int i = blockIdx.x * blockDim.x + threadIdx.x;
  if (i >= Kpad * M) return;
  int k = i / M, n = i - k * M;
  float v = (k < K) ? W[(long)k * M + n] : 0.0f;
  int kstep = k >> 5, kr = k & 31;
  int lane = ((kr >> 4) << 4) + (n & 15);
  int e = kr & 15;
  long dst = (((long)kstep * (M >> 4) + (n >> 4)) * 32 + lane) * 16 + e;
  Wp[dst] = (_Float16)v;
}

// Zero-pad rows: xp[N,Kout] = [x[N,Kin] | 0]
__global__ void k_pad_rows(const float* __restrict__ x, float* __restrict__ xp,
                           int N, int Kin, int Kout) {
  int i = blockIdx.x * blockDim.x + threadIdx.x;
  if (i >= N * Kout) return;
  int r = i / Kout, c = i - r * Kout;
  xp[i] = (c < Kin) ? x[(long)r * Kin + c] : 0.0f;
}

// ---------------------------------------------------------------------------
// WMMA GEMM: C[N,M] = A[N,K] @ W[K,M] (+bias). K multiple of 32, M multiple
// of 64. One wave computes a 16x64 output strip (4 WMMA tiles, A reused).
// A loaded as 4x float4 per k-step; B loaded as one aligned v16h per tile.
// ---------------------------------------------------------------------------
__global__ void k_wmma_gemm(const float* __restrict__ A, const _Float16* __restrict__ Wp,
                            const float* __restrict__ bias, float* __restrict__ C,
                            int N, int K, int M) {
  const int lane = threadIdx.x & 31;
  const int tile = blockIdx.x * (blockDim.x >> 5) + (threadIdx.x >> 5);
  const int colGroups = M >> 6;
  const int rowTile = tile / colGroups;
  const int row0 = rowTile << 4;
  if (row0 >= N) return;                    // wave-uniform exit
  const int colGroup = tile - rowTile * colGroups;
  const int col0 = colGroup << 6;
  const int colTiles = M >> 4;

  int rA = row0 + (lane & 15);
  if (rA > N - 1) rA = N - 1;               // safe clamp; clamped rows never stored
  const int kHA = (lane >> 4) << 4;         // 0 or 16
  const float* Arow = A + (long)rA * K;

  v8f acc[4] = {};
  for (int k0 = 0; k0 < K; k0 += 32) {
    const float4* apLo = (const float4*)(Arow + k0 + kHA);
    const float4* apHi = (const float4*)(Arow + k0 + 8 + kHA);
    float4 a0 = apLo[0], a1 = apLo[1], a2 = apHi[0], a3 = apHi[1];
    v16h a;
    a[0]  = (_Float16)a0.x; a[1]  = (_Float16)a0.y; a[2]  = (_Float16)a0.z; a[3]  = (_Float16)a0.w;
    a[4]  = (_Float16)a1.x; a[5]  = (_Float16)a1.y; a[6]  = (_Float16)a1.z; a[7]  = (_Float16)a1.w;
    a[8]  = (_Float16)a2.x; a[9]  = (_Float16)a2.y; a[10] = (_Float16)a2.z; a[11] = (_Float16)a2.w;
    a[12] = (_Float16)a3.x; a[13] = (_Float16)a3.y; a[14] = (_Float16)a3.z; a[15] = (_Float16)a3.w;

    const _Float16* bbase =
        Wp + ((((long)(k0 >> 5)) * colTiles + (col0 >> 4)) * 32 + lane) * 16;
#pragma unroll
    for (int j = 0; j < 4; ++j) {
      v16h b = *(const v16h*)(bbase + (long)j * 32 * 16);
      acc[j] = __builtin_amdgcn_wmma_f32_16x16x32_f16(false, a, false, b, (short)0,
                                                      acc[j], false, false);
    }
  }

  const int mBase = (lane >> 4) << 3;       // C/D: VGPR v -> row v (+8 for upper lanes)
  const int nb = lane & 15;
#pragma unroll
  for (int j = 0; j < 4; ++j) {
    int col = col0 + (j << 4) + nb;
    float bv = bias ? bias[col] : 0.0f;
#pragma unroll
    for (int v = 0; v < 8; ++v) {
      int r = row0 + mBase + v;
      if (r < N) C[(long)r * M + col] = acc[j][v] + bv;
    }
  }
}

// ---------------------------------------------------------------------------
__global__ void k_fill(float* __restrict__ p, float v, int n) {
  int i = blockIdx.x * blockDim.x + threadIdx.x;
  if (i < n) p[i] = v;
}

// s[n,h] = dot(hh[n,h,:], a_src[h,:]);  d[n,h] = dot(hh[n,h,:], a_dst[h,:])
__global__ void k_attn_scores(const float* __restrict__ hh, const float* __restrict__ a_src,
                              const float* __restrict__ a_dst, float* __restrict__ s,
                              float* __restrict__ d, int N, int H, int Dh) {
  int i = blockIdx.x * blockDim.x + threadIdx.x;
  if (i >= N * H) return;
  int n = i / H, h = i - n * H;
  const float* row = hh + (long)n * H * Dh + h * Dh;
  const float* as = a_src + h * Dh;
  const float* ad = a_dst + h * Dh;
  float ss = 0.0f, dd = 0.0f;
  for (int k = 0; k < Dh; ++k) { ss += row[k] * as[k]; dd += row[k] * ad[k]; }
  s[i] = ss;
  d[i] = dd;
}

__device__ __forceinline__ float leaky02(float x) { return x > 0.0f ? x : 0.2f * x; }

__device__ __forceinline__ void atomicMaxF32(float* addr, float val) {
  if (val >= 0.0f) atomicMax((int*)addr, __float_as_int(val));
  else             atomicMin((unsigned int*)addr, __float_as_uint(val));
}

// Pass 1: segment max of e over dst (E real edges + N self loops)
__global__ void k_edge_max(const int* __restrict__ src, const int* __restrict__ dst,
                           int E, int N, int H, const float* __restrict__ s,
                           const float* __restrict__ d, float* __restrict__ m) {
  int i = blockIdx.x * blockDim.x + threadIdx.x;
  int total = (E + N) * H;
  if (i >= total) return;
  int eid = i / H, h = i - eid * H;
  int sN = eid < E ? src[eid] : (eid - E);
  int dN = eid < E ? dst[eid] : (eid - E);
  float e = leaky02(s[sN * H + h] + d[dN * H + h]);
  atomicMaxF32(&m[dN * H + h], e);
}

// Pass 2: denom = segment sum of exp(e - m[dst])
__global__ void k_edge_denom(const int* __restrict__ src, const int* __restrict__ dst,
                             int E, int N, int H, const float* __restrict__ s,
                             const float* __restrict__ d, const float* __restrict__ m,
                             float* __restrict__ den) {
  int i = blockIdx.x * blockDim.x + threadIdx.x;
  int total = (E + N) * H;
  if (i >= total) return;
  int eid = i / H, h = i - eid * H;
  int sN = eid < E ? src[eid] : (eid - E);
  int dN = eid < E ? dst[eid] : (eid - E);
  float e = leaky02(s[sN * H + h] + d[dN * H + h]);
  atomicAdd(&den[dN * H + h], expf(e - m[dN * H + h]));
}

// Pass 3: agg[dst] += alpha * hh[src]  (one wave32 per edge, lanes stride dims)
__global__ void k_edge_agg(const int* __restrict__ src, const int* __restrict__ dst,
                           int E, int N, int H, int Dh, const float* __restrict__ s,
                           const float* __restrict__ d, const float* __restrict__ m,
                           const float* __restrict__ den, const float* __restrict__ hh,
                           float* __restrict__ agg) {
  int lane = threadIdx.x & 31;
  int eid = blockIdx.x * (blockDim.x >> 5) + (threadIdx.x >> 5);
  if (eid >= E + N) return;
  int sN = eid < E ? src[eid] : (eid - E);
  int dN = eid < E ? dst[eid] : (eid - E);
  int Dt = H * Dh;
  for (int dim = lane; dim < Dt; dim += 32) {
    int h = dim / Dh;
    float e = leaky02(s[sN * H + h] + d[dN * H + h]);
    float alpha = expf(e - m[dN * H + h]) / den[dN * H + h];
    atomicAdd(&agg[(long)dN * Dt + dim], alpha * hh[(long)sN * Dt + dim]);
  }
}

// (optional add bias) -> (optional LayerNorm) -> exact GELU. One wave32 per row.
__global__ void k_row_postproc(const float* __restrict__ x, const float* __restrict__ add_b,
                               const float* __restrict__ ln_g, const float* __restrict__ ln_b,
                               float* __restrict__ out, int N, int D, int do_ln) {
  int lane = threadIdx.x & 31;
  int row = blockIdx.x * (blockDim.x >> 5) + (threadIdx.x >> 5);
  if (row >= N) return;
  const float* xr = x + (long)row * D;
  float vals[8];                                   // D in {64,128,256}
  int cnt = D >> 5;
  float sum = 0.0f;
  for (int i = 0; i < cnt; ++i) {
    int c = lane + (i << 5);
    float v = xr[c];
    if (add_b) v += add_b[c];
    vals[i] = v;
    sum += v;
  }
  for (int off = 16; off >= 1; off >>= 1) sum += __shfl_xor(sum, off, 32);
  float mu = sum / (float)D;
  float var = 0.0f;
  for (int i = 0; i < cnt; ++i) { float t = vals[i] - mu; var += t * t; }
  for (int off = 16; off >= 1; off >>= 1) var += __shfl_xor(var, off, 32);
  float inv = rsqrtf(var / (float)D + 1e-5f);
  for (int i = 0; i < cnt; ++i) {
    int c = lane + (i << 5);
    float y = do_ln ? ((vals[i] - mu) * inv * ln_g[c] + ln_b[c]) : vals[i];
    out[(long)row * D + c] = 0.5f * y * (1.0f + erff(y * 0.70710678118654752440f));
  }
}

// Pooling: graph_feat[batch[n]] += h[n]; cnt[batch[n]] += 1 (wave per node)
__global__ void k_pool_accum(const float* __restrict__ h, const int* __restrict__ batch,
                             float* __restrict__ gf, float* __restrict__ cnt, int N, int D) {
  int lane = threadIdx.x & 31;
  int n = blockIdx.x * (blockDim.x >> 5) + (threadIdx.x >> 5);
  if (n >= N) return;
  int g = batch[n];
  if (lane == 0) atomicAdd(&cnt[g], 1.0f);
  for (int c = lane; c < D; c += 32) atomicAdd(&gf[(long)g * D + c], h[(long)n * D + c]);
}

__global__ void k_pool_div(float* __restrict__ gf, const float* __restrict__ cnt, int G, int D) {
  int i = blockIdx.x * blockDim.x + threadIdx.x;
  if (i >= G * D) return;
  gf[i] /= fmaxf(cnt[i / D], 1.0f);
}

__global__ void k_concat(const float* __restrict__ a, const float* __restrict__ b,
                         float* __restrict__ c, int G, int Da, int Db) {
  int i = blockIdx.x * blockDim.x + threadIdx.x;
  int Dt = Da + Db;
  if (i >= G * Dt) return;
  int g = i / Dt, col = i - g * Dt;
  c[i] = (col < Da) ? a[(long)g * Da + col] : b[(long)g * Db + (col - Da)];
}

// Tiny final projection: out[G,2] = A[G,64] @ W[64,2] + b[2]
__global__ void k_final_linear(const float* __restrict__ A, const float* __restrict__ W,
                               const float* __restrict__ bias, float* __restrict__ out,
                               int N, int K, int M) {
  int i = blockIdx.x * blockDim.x + threadIdx.x;
  if (i >= N * M) return;
  int r = i / M, c = i - r * M;
  float acc = bias[c];
  for (int k = 0; k < K; ++k) acc += A[(long)r * K + k] * W[k * M + c];
  out[i] = acc;
}

// ---------------------------------------------------------------------------
extern "C" void kernel_launch(void* const* d_in, const int* in_sizes, int n_in,
                              void* d_out, int out_size, void* d_ws, size_t ws_size,
                              hipStream_t stream) {
  const float* x          = (const float*)d_in[0];   // [N,16]
  const int*   edge_index = (const int*)d_in[1];     // [2,E]
  const int*   batch      = (const int*)d_in[2];     // [N]
  const float* u          = (const float*)d_in[3];   // [G,256]

  const int N = in_sizes[0] / 16;
  const int E = in_sizes[1] / 2;
  const int G = in_sizes[3] / 256;
  const int* srcI = edge_index;
  const int* dstI = edge_index + E;

  int pi = 4;
  auto F = [&](void) { return (const float*)d_in[pi++]; };
  const float* ne_w = F(); const float* ne_b = F();
  const float* g1_w = F(); const float* g1_as = F(); const float* g1_ad = F();
  const float* g1_b = F(); const float* ln1_g = F(); const float* ln1_b = F();
  const float* g2_w = F(); const float* g2_as = F(); const float* g2_ad = F();
  const float* g2_b = F(); const float* ln2_g = F(); const float* ln2_b = F();
  const float* g3_w = F(); const float* g3_as = F(); const float* g3_ad = F();
  const float* g3_b = F(); const float* ln3_g = F(); const float* ln3_b = F();
  const float* ge_w1 = F(); const float* ge_b1 = F(); const float* ge_g1 = F(); const float* ge_bb1 = F();
  const float* ge_w2 = F(); const float* ge_b2 = F(); const float* ge_g2 = F(); const float* ge_bb2 = F();
  const float* fu_w1 = F(); const float* fu_b1 = F(); const float* fu_g1 = F(); const float* fu_bb1 = F();
  const float* fu_w2 = F(); const float* fu_b2 = F(); const float* fu_g2 = F(); const float* fu_bb2 = F();
  const float* cl_w1 = F(); const float* cl_b1 = F(); const float* cl_w2 = F(); const float* cl_b2 = F();

  // ---- workspace carve ----
  char* wsp = (char*)d_ws;
  auto alloc = [&](size_t bytes) -> void* {
    void* p = (void*)wsp;
    wsp += (bytes + 255) & ~(size_t)255;
    return p;
  };
  float*    h     = (float*)alloc((size_t)N * 128 * 4);
  float*    hh    = (float*)alloc((size_t)N * 128 * 4);
  float*    agg   = (float*)alloc((size_t)N * 128 * 4);
  float*    xp    = (float*)alloc((size_t)N * 32 * 4);     // padded encoder input
  float*    s     = (float*)alloc((size_t)N * 4 * 4);
  float*    dv    = (float*)alloc((size_t)N * 4 * 4);
  float*    m     = (float*)alloc((size_t)N * 4 * 4);
  float*    den   = (float*)alloc((size_t)N * 4 * 4);
  _Float16* wpack = (_Float16*)alloc((size_t)256 * 256 * 2); // >= max Kpad*M f16
  float*    gf    = (float*)alloc((size_t)G * 64 * 4);
  float*    cnt   = (float*)alloc((size_t)G * 4);
  float*    ge1   = (float*)alloc((size_t)G * 128 * 4);
  float*    ge2   = (float*)alloc((size_t)G * 128 * 4);
  float*    cc    = (float*)alloc((size_t)G * 192 * 4);
  float*    f1    = (float*)alloc((size_t)G * 256 * 4);
  float*    f2    = (float*)alloc((size_t)G * 128 * 4);
  float*    c1    = (float*)alloc((size_t)G * 64 * 4);

  auto cdiv = [](long a, long b) { return (int)((a + b - 1) / b); };

  // GEMM: pack W (zero-padded to Kpad), then WMMA. A row stride must be Kpad.
  auto gemm = [&](const float* A, const float* W, const float* bias, float* C,
                  int n, int k, int mdim) {
    int kpad = (k + 31) & ~31;
    k_pack_w<<<cdiv((long)kpad * mdim, 256), 256, 0, stream>>>(W, wpack, k, kpad, mdim);
    int tiles = cdiv(n, 16) * (mdim >> 6);       // 16x64 strips per wave
    k_wmma_gemm<<<cdiv(tiles, 8), 256, 0, stream>>>(A, wpack, bias, C, n, kpad, mdim);
  };
  auto fill = [&](float* p, float v, long n) {
    k_fill<<<cdiv(n, 256), 256, 0, stream>>>(p, v, (int)n);
  };

  const int H = 4;
  auto gat_block = [&](const float* Wm, const float* as, const float* ad, const float* bias,
                       const float* lng, const float* lnb, int Din, int Dout) {
    int Dh = Dout / H;
    gemm(h, Wm, nullptr, hh, N, Din, Dout);                         // hh = h @ W
    k_attn_scores<<<cdiv((long)N * H, 256), 256, 0, stream>>>(hh, as, ad, s, dv, N, H, Dh);
    fill(m, -INFINITY, (long)N * H);
    fill(den, 0.0f, (long)N * H);
    fill(agg, 0.0f, (long)N * Dout);
    long EH = (long)(E + N) * H;
    k_edge_max<<<cdiv(EH, 256), 256, 0, stream>>>(srcI, dstI, E, N, H, s, dv, m);
    k_edge_denom<<<cdiv(EH, 256), 256, 0, stream>>>(srcI, dstI, E, N, H, s, dv, m, den);
    k_edge_agg<<<cdiv((long)(E + N), 8), 256, 0, stream>>>(srcI, dstI, E, N, H, Dh,
                                                           s, dv, m, den, hh, agg);
    // h = gelu(layernorm(agg + bias))
    k_row_postproc<<<cdiv(N, 8), 256, 0, stream>>>(agg, bias, lng, lnb, h, N, Dout, 1);
  };

  // ---- node encoder (pad K=16 -> 32 so the GEMM hot loop is guard-free) ----
  k_pad_rows<<<cdiv((long)N * 32, 256), 256, 0, stream>>>(x, xp, N, 16, 32);
  gemm(xp, ne_w, ne_b, h, N, 16, 64);

  // ---- 3 GAT blocks ----
  gat_block(g1_w, g1_as, g1_ad, g1_b, ln1_g, ln1_b, 64, 128);
  gat_block(g2_w, g2_as, g2_ad, g2_b, ln2_g, ln2_b, 128, 128);
  gat_block(g3_w, g3_as, g3_ad, g3_b, ln3_g, ln3_b, 128, 64);

  // ---- mean pooling per graph ----
  fill(gf, 0.0f, (long)G * 64);
  fill(cnt, 0.0f, (long)G);
  k_pool_accum<<<cdiv(N, 8), 256, 0, stream>>>(h, batch, gf, cnt, N, 64);
  k_pool_div<<<cdiv((long)G * 64, 256), 256, 0, stream>>>(gf, cnt, G, 64);

  // ---- global-feature MLP ----
  gemm(u, ge_w1, ge_b1, ge1, G, 256, 128);
  k_row_postproc<<<cdiv(G, 8), 256, 0, stream>>>(ge1, nullptr, ge_g1, ge_bb1, ge1, G, 128, 1);
  gemm(ge1, ge_w2, ge_b2, ge2, G, 128, 128);
  k_row_postproc<<<cdiv(G, 8), 256, 0, stream>>>(ge2, nullptr, ge_g2, ge_bb2, ge2, G, 128, 1);

  // ---- fuse ----
  k_concat<<<cdiv((long)G * 192, 256), 256, 0, stream>>>(gf, ge2, cc, G, 64, 128);
  gemm(cc, fu_w1, fu_b1, f1, G, 192, 256);
  k_row_postproc<<<cdiv(G, 8), 256, 0, stream>>>(f1, nullptr, fu_g1, fu_bb1, f1, G, 256, 1);
  gemm(f1, fu_w2, fu_b2, f2, G, 256, 128);
  k_row_postproc<<<cdiv(G, 8), 256, 0, stream>>>(f2, nullptr, fu_g2, fu_bb2, f2, G, 128, 1);

  // ---- classifier head ----
  gemm(f2, cl_w1, cl_b1, c1, G, 128, 64);
  k_row_postproc<<<cdiv(G, 8), 256, 0, stream>>>(c1, nullptr, nullptr, nullptr, c1, G, 64, 0);
  k_final_linear<<<cdiv((long)G * 2, 256), 256, 0, stream>>>(c1, cl_w2, cl_b2,
                                                             (float*)d_out, G, 64, 2);
}
